// GE2ELoss_12850542150213
// MI455X (gfx1250) — matrix operand order repaired and checked by reading, not compiled, not run on hardware
//
#include <hip/hip_runtime.h>
#include <hip/hip_bf16.h>

#define N_PLAYERS 2048
#define M_GAMES   16
#define D_EMB     64
#define EPS       1e-8f
#define LOG2E_F   1.4426950408889634f
#define LN2_F     0.6931471805599453f
#define NTILES    (N_PLAYERS / 16)   // 128 column tiles
#define WAVES_PB  16                 // waves per workgroup (one row tile each)
#define STAGE_T   4                  // column tiles per LDS stage (8 KB)
#define NSTAGES   (NTILES / STAGE_T) // 32 double-buffered stages

typedef __attribute__((ext_vector_type(16))) __bf16 v16bf;
typedef __attribute__((ext_vector_type(8)))  __bf16 v8bf;
typedef __attribute__((ext_vector_type(8)))  float  v8f;

__global__ void ge2e_zero_out(float* out) { out[0] = 0.0f; }

// One wave per player: leave-one-out centroid, L2-normalize, emit bf16 copies
// of gv and centroids in [M][N][D] layout + per-(game,player) row norms
// (Cauchy-Schwarz bound for the max-free logsumexp in the main kernel).
__global__ __launch_bounds__(32) void ge2e_prep_kernel(
    const float* __restrict__ gv, __bf16* __restrict__ gvb,
    __bf16* __restrict__ cenb, float* __restrict__ gnorm) {
  const int i    = blockIdx.x;
  const int lane = threadIdx.x;
  const int d0   = lane * 2;                      // 2 dims per lane (64/32)
  const float* rowbase = gv + (size_t)i * M_GAMES * D_EMB;

  float s0 = 0.0f, s1 = 0.0f;
#pragma unroll
  for (int j = 0; j < M_GAMES; ++j) {
    s0 += rowbase[j * D_EMB + d0];
    s1 += rowbase[j * D_EMB + d0 + 1];
  }
  const float inv_m1 = 1.0f / (float)(M_GAMES - 1);
#pragma unroll
  for (int j = 0; j < M_GAMES; ++j) {
    const float g0 = rowbase[j * D_EMB + d0];
    const float g1 = rowbase[j * D_EMB + d0 + 1];
    const float l0 = (s0 - g0) * inv_m1;          // leave-one-out mean
    const float l1 = (s1 - g1) * inv_m1;
    float p = l0 * l0 + l1 * l1;                  // centroid norm^2
    float q = g0 * g0 + g1 * g1;                  // raw game-vector norm^2
#pragma unroll
    for (int msk = 16; msk >= 1; msk >>= 1) {
      p += __shfl_xor(p, msk, 32);
      q += __shfl_xor(q, msk, 32);
    }
    const float inv = 1.0f / (sqrtf(p) + EPS);
    const size_t o = ((size_t)j * N_PLAYERS + i) * D_EMB + d0;
    gvb[o]      = (__bf16)g0;
    gvb[o + 1]  = (__bf16)g1;
    cenb[o]     = (__bf16)(l0 * inv);
    cenb[o + 1] = (__bf16)(l1 * inv);
    if (lane == 0) gnorm[(size_t)j * N_PLAYERS + i] = sqrtf(q);
  }
}

// Cooperative GEMM + fused max-free logsumexp.
// Workgroup = 16 waves; wave w owns one 16-row A tile. Column tiles are
// staged 4-at-a-time (8 KB) in double-buffered LDS via
// global_load_async_to_lds_b128 (ASYNCcnt): every wave DMAs one 512 B slice
// and waits on its own counter, so 2 barriers cover 8 WMMAs and the
// barrier-free unrolled inner loop lets ds_loads overlap WMMAs/exp.
__global__ __launch_bounds__(32 * WAVES_PB) void ge2e_main_kernel(
    const __bf16* __restrict__ gvb, const __bf16* __restrict__ cenb,
    const float* __restrict__ gnorm, const float* __restrict__ wp,
    float* __restrict__ out) {
  __shared__ v16bf lds_B[2][STAGE_T][16][4];   // 2 x 8 KB staging buffers

  const int tb   = blockIdx.x & 7;                       // 8 blocks per game
  const int j    = blockIdx.x >> 3;                      // game index
  const int lane = threadIdx.x & 31;
  const int wid  = __builtin_amdgcn_readfirstlane(threadIdx.x >> 5);
  const int tile = __builtin_amdgcn_readfirstlane(tb * WAVES_PB + wid);
  const int nidx = lane & 15;
  const int half = lane >> 4;
  const float w   = wp[0];
  const float wl2 = w * LOG2E_F;

  const __bf16* Gj = gvb  + (size_t)j * N_PLAYERS * D_EMB;
  const __bf16* Cj = cenb + (size_t)j * N_PLAYERS * D_EMB;

  // A fragments (16x32 bf16): lane holds row M = lane&15;
  // lanes 0-15 carry K = 0..15, lanes 16-31 carry K = 16..31.
  const __bf16* arow = Gj + (size_t)(tile * 16 + nidx) * D_EMB;
  const v8bf p00 = *(const v8bf*)(arow +      half * 16);
  const v8bf p01 = *(const v8bf*)(arow +  8 + half * 16);
  const v8bf p10 = *(const v8bf*)(arow + 32 + half * 16);
  const v8bf p11 = *(const v8bf*)(arow + 40 + half * 16);
  const v16bf a0 = __builtin_shufflevector(p00, p01, 0,1,2,3,4,5,6,7,8,9,10,11,12,13,14,15);
  const v16bf a1 = __builtin_shufflevector(p10, p11, 0,1,2,3,4,5,6,7,8,9,10,11,12,13,14,15);

  // Per-slot exp2-domain bias: slot r covers row tile*16 + r + 8*half.
  const float* gn = gnorm + (size_t)j * N_PLAYERS + tile * 16;
  float wn[8], ssum[8], corr[8];
#pragma unroll
  for (int r = 0; r < 8; ++r) {
    wn[r]   = gn[r + 8 * half] * wl2;   // w*log2e*||gv_row||
    ssum[r] = 0.0f;
    corr[r] = 0.0f;
  }

  // Async DMA of one 8 KB stage (4 column tiles): 16 waves x 512 B slices,
  // one global_load_async_to_lds_b128 per wave (lane granularity 16 B).
  const unsigned char* Cbytes = (const unsigned char*)Cj;
  auto issue_stage = [&](int st, int buf) {
    unsigned loff = (unsigned)(uintptr_t)(&lds_B[buf][0][0][0]) +
                    (unsigned)(wid * 512 + lane * 16);
    const void* gsrc = Cbytes + (size_t)st * 8192 + wid * 512 + lane * 16;
    asm volatile("global_load_async_to_lds_b128 %0, %1, off"
                 :: "v"(loff), "v"(gsrc) : "memory");
  };

  issue_stage(0, 0);                      // prologue: prefetch first stage

  for (int s = 0; s < NSTAGES; ++s) {
    const int p = s & 1;
    if (s + 1 < NSTAGES) {
      issue_stage(s + 1, p ^ 1);          // overlap next stage's DMA
      asm volatile("s_wait_asynccnt 0x1" ::: "memory");  // my old slice done
    } else {
      asm volatile("s_wait_asynccnt 0x0" ::: "memory");
    }
    __syncthreads();                      // publish stage p to all waves

#pragma unroll
    for (int t = 0; t < STAGE_T; ++t) {   // barrier-free: ds/WMMA/exp overlap
      const int kk = s * STAGE_T + t;
      // B fragment (32x16 bf16) from LDS: lane = column nidx; lanes 0-15
      // carry K = 0..15, lanes 16-31 carry K = 16..31 -> 32 B per fragment.
      const v16bf b0 = lds_B[p][t][nidx][half];
      const v16bf b1 = lds_B[p][t][nidx][2 + half];

      v8f c = {};
      c = __builtin_amdgcn_wmma_f32_16x16x32_bf16(false, a0, false, b0,
                                                  (short)0, c, false, false);
      c = __builtin_amdgcn_wmma_f32_16x16x32_bf16(false, a1, false, b1,
                                                  (short)0, c, false, false);

      // Max-free accumulation: ssum[r] += 2^(dot*w*log2e - w*log2e*||row||).
#pragma unroll
      for (int r = 0; r < 8; ++r)
        ssum[r] += __builtin_amdgcn_exp2f(fmaf(c[r], wl2, -wn[r]));

      if (kk == tile) {                   // scalar-uniform branch: taken once
#pragma unroll
        for (int r = 0; r < 8; ++r)
          if (nidx == (r + 8 * half)) corr[r] = c[r];  // raw diagonal dot
      }
    }
    __syncthreads();                      // retire stage p before overwrite
  }

  // Sum partials across the 16 lanes of each half group (pure adds: no max
  // merge needed thanks to the fixed bound). b cancels between lse and corr.
  float local = 0.0f;
#pragma unroll
  for (int r = 0; r < 8; ++r) {
    float s = ssum[r], cr = corr[r];
#pragma unroll
    for (int msk = 1; msk < 16; msk <<= 1) {
      s  += __shfl_xor(s,  msk, 32);
      cr += __shfl_xor(cr, msk, 32);
    }
    // lse - corr = w*||row|| + ln(s) - w*dot_diag
    local += wn[r] * LN2_F + __builtin_amdgcn_logf(s) * LN2_F - w * cr;
  }
  local = (nidx == 0) ? local : 0.0f;     // deduplicate 16x replication
  local += __shfl_xor(local, 16, 32);     // combine the two half groups
  if (lane == 0)
    atomicAdd(out, local * (1.0f / (float)(N_PLAYERS * M_GAMES)));
}

extern "C" void kernel_launch(void* const* d_in, const int* in_sizes, int n_in,
                              void* d_out, int out_size, void* d_ws, size_t ws_size,
                              hipStream_t stream) {
  const float* gv = (const float*)d_in[0];
  const float* w  = (const float*)d_in[1];
  // d_in[2] (b) cancels exactly between lse and the diagonal term.
  // d_in[3] (player_indices) encodes the balanced sorted layout assumed here.
  float* out = (float*)d_out;

  __bf16* gvb   = (__bf16*)d_ws;                              // [M][N][D] bf16, 4 MB
  __bf16* cenb  = gvb + (size_t)N_PLAYERS * M_GAMES * D_EMB;  // [M][N][D] bf16, 4 MB
  float*  gnorm = (float*)(cenb + (size_t)N_PLAYERS * M_GAMES * D_EMB);  // 128 KB

  ge2e_zero_out<<<1, 1, 0, stream>>>(out);
  ge2e_prep_kernel<<<N_PLAYERS, 32, 0, stream>>>(gv, gvb, cenb, gnorm);
  // 16 games x 8 row-tile groups = 128 workgroups of 16 waves.
  ge2e_main_kernel<<<M_GAMES * (NTILES / WAVES_PB), 32 * WAVES_PB, 0, stream>>>(
      gvb, cenb, gnorm, w, out);
}
